// GNN_74311524155560
// MI455X (gfx1250) — compile-verified, matrix operand order checked by reading, not run on hardware
//
#include <hip/hip_runtime.h>

// ---------------------------------------------------------------------------
// GIN + virtual-node GNN forward for MI455X (gfx1250, wave32, WMMA + TDM).
// Dense GEMMs: v_wmma_f32_16x16x32_bf16, A tiles streamed to LDS by the
// Tensor Data Mover (tensor_load_to_lds, TENSORcnt), double buffered.
// ---------------------------------------------------------------------------

typedef __bf16 bf16;
typedef __attribute__((ext_vector_type(16))) __bf16 v16bf;
typedef __attribute__((ext_vector_type(8)))  __bf16 v8bf;
typedef __attribute__((ext_vector_type(4)))  __bf16 v4bf;
typedef __attribute__((ext_vector_type(8)))  float   v8f;
typedef __attribute__((ext_vector_type(4)))  unsigned int uint32x4;
typedef __attribute__((ext_vector_type(8)))  int          int32x8;
typedef __attribute__((ext_vector_type(4)))  int          int32x4;

#define BN_SCALE 0.99999500003749972f  // 1/sqrt(1+1e-5)

static constexpr int Nn = 30000, Ee = 400000, Ll = 5, Dd = 300, Gg = 1000, Cc = 10;
// GEMM1: K=300 -> pad 320, Nout=600 -> pad 640
static constexpr int K1 = 300, N1 = 600, K1p = 320, N1p = 640, KT1 = 10;
// GEMM2: K=600 -> pad 640, Nout=300 -> pad 320
static constexpr int K2 = 600, N2 = 300, K2p = 640, N2p = 320, KT2 = 20;
static constexpr int PACKSZ = 204800;   // halfs per packed weight matrix (both shapes)
static constexpr int LDS_STRIDE = 40;   // halfs per LDS row (32 + 8 pad, via TDM padding)

static inline int cdiv(int a, int b) { return (a + b - 1) / b; }

// K permutation inside a 16-element fragment chunk (matches documented A layout:
// lanes 0-15 hold K=[h*8 .. h*8+7] then [16+h*8 .. 16+h*8+7], h = lane>>4).
__device__ __forceinline__ int kmap(int e, int h) {
  return (e < 8) ? (h * 8 + e) : (16 + h * 8 + (e - 8));
}

// ---------------------------------------------------------------------------
// Weight repack: fp32 row-major [K x Nout] -> bf16 WMMA B-fragment layout.
// ---------------------------------------------------------------------------
__global__ void k_pack(const float* __restrict__ W, int K, int Nout, int KT,
                       int total, bf16* __restrict__ Bp) {
  int idx = blockIdx.x * 256 + threadIdx.x;
  if (idx >= total) return;
  int e    = idx & 15;
  int lane = (idx >> 4) & 31;
  int frag = idx >> 9;
  int kc   = frag % KT;
  int nt   = frag / KT;
  int k = (kc << 5) + kmap(e, lane >> 4);
  int n = (nt << 4) + (lane & 15);
  float v = (k < K && n < Nout) ? W[(size_t)k * Nout + n] : 0.0f;
  Bp[idx] = (bf16)v;
}

// ---------------------------------------------------------------------------
// TDM: DMA one 128-row x 32-half bf16 tile (row stride Kpad halfs in memory)
// into LDS at byte offset ldsOff, with 16B of padding every 64B row so the
// LDS row stride becomes 40 halfs. rowsAvail rows are valid; the TDM's own
// tensor_dim1 OOB handling zero-fills rows past the end of the matrix.
// D# bitfields per CDNA5 ISA ch. 8 (group0 128b, group1 256b; 2D tensor so
// the remaining descriptor groups are zero). 6-arg builtin form.
// ---------------------------------------------------------------------------
__device__ __forceinline__ void tdm_load_tile_128x32(const bf16* g, unsigned ldsOff,
                                                     int rowsAvail, int kpad) {
  unsigned long long ga = (unsigned long long)(size_t)g;
  unsigned rows = (rowsAvail < 0) ? 0u : (unsigned)rowsAvail;

  uint32x4 g0;
  g0[0] = 1u;                                      // count=1 (valid), no gather
  g0[1] = ldsOff;                                  // lds_addr (bytes)
  g0[2] = (unsigned)(ga & 0xffffffffu);            // global_addr[31:0]
  g0[3] = (unsigned)((ga >> 32) & 0x01ffffffu)     // global_addr[56:32]
        | (2u << 30);                              // type=2 ("image")

  int32x8 g1;
  g1[0] = (int)((1u << 16)                         // data_size=1 -> 2 bytes
              | (1u << 20)                         // pad_enable
              | (3u << 22)                         // pad_interval: every 16 DWORDs
              | (3u << 25));                       // pad_amount: 4 DWORDs (16B)
  g1[1] = (int)(((unsigned)kpad & 0xffffu) << 16); // tensor_dim0 lo16 (bits 63:48)
  g1[2] = (int)((((unsigned)kpad >> 16) & 0xffffu) // tensor_dim0 hi16 (bits 79:64)
              | ((rows & 0xffffu) << 16));         // tensor_dim1 lo16 (bits 95:80)
  g1[3] = (int)(((rows >> 16) & 0xffffu)           // tensor_dim1 hi16 (bits 111:96)
              | (32u << 16));                      // tile_dim0 = 32 (bits 127:112)
  g1[4] = (int)128u;                               // tile_dim1 = 128, tile_dim2 = 0
  g1[5] = (int)(unsigned)kpad;                     // tensor_dim0_stride lo32
  g1[6] = 0;                                       // stride hi16, dim1_stride lo16
  g1[7] = 0;

  int32x4 z4 = {0, 0, 0, 0};
  int32x8 z8 = {0, 0, 0, 0, 0, 0, 0, 0};
  __builtin_amdgcn_tensor_load_to_lds(g0, g1, z4, z4, z8, 0);
}

// ---------------------------------------------------------------------------
// WMMA GEMM: C[M, Nout] = A[M, Kpad](bf16) x Bpack, fused bias + eval-BN (+ReLU).
// 256 threads = 8 waves; tile 128(M) x 64(N); wave owns 2x2 16x16 accumulators.
// A tiles arrive via TDM into double-buffered LDS; B fragments are coalesced
// global_load_b128 pairs from the packed weights (L2 resident, 400KB).
// ---------------------------------------------------------------------------
template <int STORE_BF16>
__global__ __launch_bounds__(256) void gemm_wmma(
    const bf16* __restrict__ A, int M, int Kpad,
    const bf16* __restrict__ Bpack, int NoutReal, int NoutPad,
    const float* __restrict__ bias, const float* __restrict__ gamma,
    const float* __restrict__ beta, int do_relu,
    bf16* __restrict__ outBf, float* __restrict__ outF) {
  __shared__ bf16 As[2][128 * LDS_STRIDE];

  const int tid   = threadIdx.x;
  const int lane  = tid & 31;
  const int wave  = tid >> 5;
  const int waveM = wave & 3;   // 0..3 -> 32-row slice
  const int waveN = wave >> 2;  // 0..1 -> 32-col slice
  const int lrow  = lane & 15;
  const int lhalf = lane >> 4;
  const int blockM0 = blockIdx.x * 128;
  const int blockN0 = blockIdx.y * 64;
  const int KT = Kpad >> 5;
  const int nt0 = blockIdx.y * 4 + waveN * 2;  // first global 16-col tile of wave

  v8f acc[2][2] = {};

  const bf16* tileBase = A + (size_t)blockM0 * Kpad;
  const int rowsAvail = M - blockM0;
  const unsigned lds0 = (unsigned)(size_t)&As[0][0];
  const unsigned lds1 = (unsigned)(size_t)&As[1][0];

  // Prologue: wave 0 kicks off the DMA of tile 0.
  if (wave == 0) {
    tdm_load_tile_128x32(tileBase, lds0, rowsAvail, Kpad);
  }

  for (int kc = 0; kc < KT; ++kc) {
    if (wave == 0) {
      if (kc + 1 < KT) {
        // Issue next tile, then wait until the *previous* one has landed
        // (TDM ops from one wave complete in order).
        tdm_load_tile_128x32(tileBase + ((kc + 1) << 5),
                             ((kc + 1) & 1) ? lds1 : lds0, rowsAvail, Kpad);
        __builtin_amdgcn_s_wait_tensorcnt((short)1);
      } else {
        __builtin_amdgcn_s_wait_tensorcnt((short)0);
      }
    }
    if (kc + 1 < KT) {  // warm L2 for next K-step's B fragments
      __builtin_prefetch(Bpack + ((size_t)(nt0 * KT + kc + 1) << 9) + (lane << 4), 0, 1);
    }
    __syncthreads();  // tile kc visible to all waves

    const bf16* buf = (kc & 1) ? &As[1][0] : &As[0][0];

    // A fragments from LDS: two ds_load_b128 per 16x32 tile per lane
    v16bf afrag[2];
#pragma unroll
    for (int mt = 0; mt < 2; ++mt) {
      const bf16* p = buf + (waveM * 32 + mt * 16 + lrow) * LDS_STRIDE + lhalf * 8;
      v8bf lo = *(const v8bf*)p;
      v8bf hi = *(const v8bf*)(p + 16);
      afrag[mt] = __builtin_shufflevector(lo, hi, 0, 1, 2, 3, 4, 5, 6, 7,
                                          8, 9, 10, 11, 12, 13, 14, 15);
    }
#pragma unroll
    for (int nt = 0; nt < 2; ++nt) {
      const bf16* bp = Bpack + ((size_t)((nt0 + nt) * KT + kc) << 9) + (lane << 4);
      v16bf bfrag = *(const v16bf*)bp;  // 2x global_load_b128, fully coalesced
#pragma unroll
      for (int mt = 0; mt < 2; ++mt) {
        acc[mt][nt] = __builtin_amdgcn_wmma_f32_16x16x32_bf16(
            false, afrag[mt], false, bfrag, (short)0, acc[mt][nt], false, false);
      }
    }
    __syncthreads();  // everyone done reading before this buffer is re-filled
  }

  // Epilogue: u = acc + bias; v = gamma*(u*BN_SCALE) + beta; optional ReLU.
#pragma unroll
  for (int nt = 0; nt < 2; ++nt) {
    const int col = blockN0 + waveN * 32 + nt * 16 + lrow;
    const bool cok = (col < NoutReal);
    const float bi = cok ? bias[col] : 0.0f;
    const float gg = cok ? gamma[col] : 0.0f;
    const float bb = cok ? beta[col] : 0.0f;
#pragma unroll
    for (int mt = 0; mt < 2; ++mt) {
#pragma unroll
      for (int r = 0; r < 8; ++r) {
        const int row = blockM0 + waveM * 32 + mt * 16 + (lhalf << 3) + r;
        if (row >= M) continue;
        float v = 0.0f;
        if (cok) {
          v = gg * ((acc[mt][nt][r] + bi) * BN_SCALE) + bb;
          if (do_relu) v = fmaxf(v, 0.0f);
        }
        if (STORE_BF16) {
          // also zero-fills pad columns [NoutReal, NoutPad) for the next GEMM
          outBf[(size_t)row * NoutPad + col] = (bf16)v;
        } else if (cok) {
          outF[(size_t)row * NoutReal + col] = v;
        }
      }
    }
  }
}

// ---------------------------------------------------------------------------
// Elementwise / scatter kernels
// ---------------------------------------------------------------------------
__global__ void k_init_vn(const float* __restrict__ emb, float* __restrict__ vn,
                          int total) {
  int i = blockIdx.x * 256 + threadIdx.x;
  if (i < total) vn[i] = emb[i % Dd];
}

// hb = h + vn[batch]   (float4 over N x 300)
__global__ void k_vn_add(const float* __restrict__ h, const float* __restrict__ vn,
                         const int* __restrict__ batch, float* __restrict__ hb) {
  int i = blockIdx.x * 256 + threadIdx.x;
  if (i >= Nn * 75) return;
  int row = i / 75, c = (i % 75) * 4;
  int g = batch[row];
  const float4 hv = *(const float4*)(h + (size_t)row * Dd + c);
  const float4 vv = *(const float4*)(vn + (size_t)g * Dd + c);
  float4 o;
  o.x = hv.x + vv.x; o.y = hv.y + vv.y; o.z = hv.z + vv.z; o.w = hv.w + vv.w;
  *(float4*)(hb + (size_t)row * Dd + c) = o;
}

// agg[dst] += relu(hb[src] + edge_attr)   (float4 per thread + 4 f32 atomics)
__global__ __launch_bounds__(256) void k_edge(
    const float* __restrict__ hb, const float* __restrict__ ea,
    const int* __restrict__ src, const int* __restrict__ dst,
    float* __restrict__ agg) {
  unsigned i = blockIdx.x * 256u + threadIdx.x;
  if (i >= (unsigned)Ee * 75u) return;
  unsigned e = i / 75u;
  int c = (int)(i % 75u) * 4;
  int s = src[e], d = dst[e];
  const float4 hv = *(const float4*)(hb + (size_t)s * Dd + c);
  const float4 ev = *(const float4*)(ea + (size_t)e * Dd + c);
  float4 m;
  m.x = fmaxf(hv.x + ev.x, 0.0f);
  m.y = fmaxf(hv.y + ev.y, 0.0f);
  m.z = fmaxf(hv.z + ev.z, 0.0f);
  m.w = fmaxf(hv.w + ev.w, 0.0f);
  float* pp = agg + (size_t)d * Dd + c;
  atomicAdd(pp + 0, m.x);
  atomicAdd(pp + 1, m.y);
  atomicAdd(pp + 2, m.z);
  atomicAdd(pp + 3, m.w);
}

// Abf = bf16((1+eps)*hb + agg), K-padded to 320 with zeros
__global__ void k_make_A1(const float* __restrict__ hb, const float* __restrict__ agg,
                          const float* __restrict__ epsP, bf16* __restrict__ Ab) {
  int i = blockIdx.x * 256 + threadIdx.x;
  if (i >= Nn * 80) return;
  int row = i / 80, c = (i % 80) * 4;
  v4bf o;
  if (c < Dd) {
    float ep = 1.0f + epsP[0];
    const float4 hv = *(const float4*)(hb + (size_t)row * Dd + c);
    const float4 av = *(const float4*)(agg + (size_t)row * Dd + c);
    o[0] = (bf16)(ep * hv.x + av.x);
    o[1] = (bf16)(ep * hv.y + av.y);
    o[2] = (bf16)(ep * hv.z + av.z);
    o[3] = (bf16)(ep * hv.w + av.w);
  } else {
    o[0] = (bf16)0.0f; o[1] = (bf16)0.0f; o[2] = (bf16)0.0f; o[3] = (bf16)0.0f;
  }
  *(v4bf*)(Ab + (size_t)row * K1p + c) = o;
}

// pooledbf = bf16(pooled + vn), K-padded to 320
__global__ void k_make_poolA(const float* __restrict__ pooled,
                             const float* __restrict__ vn, bf16* __restrict__ Ab) {
  int i = blockIdx.x * 256 + threadIdx.x;
  if (i >= Gg * 80) return;
  int row = i / 80, c = (i % 80) * 4;
  v4bf o;
  if (c < Dd) {
    const float4 pv = *(const float4*)(pooled + (size_t)row * Dd + c);
    const float4 vv = *(const float4*)(vn + (size_t)row * Dd + c);
    o[0] = (bf16)(pv.x + vv.x);
    o[1] = (bf16)(pv.y + vv.y);
    o[2] = (bf16)(pv.z + vv.z);
    o[3] = (bf16)(pv.w + vv.w);
  } else {
    o[0] = (bf16)0.0f; o[1] = (bf16)0.0f; o[2] = (bf16)0.0f; o[3] = (bf16)0.0f;
  }
  *(v4bf*)(Ab + (size_t)row * K1p + c) = o;
}

// pooled[batch[row]] += v[row]   (segment sum via atomics)
__global__ void k_segsum(const float* __restrict__ v, const int* __restrict__ batch,
                         float* __restrict__ pooled) {
  int i = blockIdx.x * 256 + threadIdx.x;
  if (i >= Nn * 75) return;
  int row = i / 75, c = (i % 75) * 4;
  int g = batch[row];
  const float4 hv = *(const float4*)(v + (size_t)row * Dd + c);
  float* pp = pooled + (size_t)g * Dd + c;
  atomicAdd(pp + 0, hv.x);
  atomicAdd(pp + 1, hv.y);
  atomicAdd(pp + 2, hv.z);
  atomicAdd(pp + 3, hv.w);
}

__global__ void k_counts(const int* __restrict__ batch, float* __restrict__ counts) {
  int i = blockIdx.x * 256 + threadIdx.x;
  if (i < Nn) atomicAdd(counts + batch[i], 1.0f);
}

// out[g,c] = (sum_h[g] / max(count,1)) . head_w[:,c] + head_b[c]
__global__ void k_head(const float* __restrict__ poolH, const float* __restrict__ counts,
                       const float* __restrict__ hw, const float* __restrict__ hbias,
                       float* __restrict__ out) {
  int i = blockIdx.x * 256 + threadIdx.x;
  if (i >= Gg * Cc) return;
  int g = i / Cc, c = i % Cc;
  float inv = 1.0f / fmaxf(counts[g], 1.0f);
  float acc = 0.0f;
  for (int k = 0; k < Dd; ++k)
    acc += poolH[(size_t)g * Dd + k] * hw[(size_t)k * Cc + c];
  out[i] = acc * inv + hbias[c];
}

// ---------------------------------------------------------------------------
// Host orchestration
// ---------------------------------------------------------------------------
extern "C" void kernel_launch(void* const* d_in, const int* in_sizes, int n_in,
                              void* d_out, int out_size, void* d_ws, size_t ws_size,
                              hipStream_t stream) {
  const float* x         = (const float*)d_in[0];
  const float* edge_attr = (const float*)d_in[1];
  const int*   eidx      = (const int*)d_in[2];
  const int*   batch     = (const int*)d_in[3];
  const float* conv_w1   = (const float*)d_in[4];
  const float* conv_b1   = (const float*)d_in[5];
  const float* conv_bng  = (const float*)d_in[6];
  const float* conv_bnb  = (const float*)d_in[7];
  const float* conv_w2   = (const float*)d_in[8];
  const float* conv_b2   = (const float*)d_in[9];
  const float* conv_eps  = (const float*)d_in[10];
  const float* bn_g      = (const float*)d_in[11];
  const float* bn_b      = (const float*)d_in[12];
  const float* vn_emb    = (const float*)d_in[13];
  const float* vn_w1     = (const float*)d_in[14];
  const float* vn_b1     = (const float*)d_in[15];
  const float* vn_bn1g   = (const float*)d_in[16];
  const float* vn_bn1b   = (const float*)d_in[17];
  const float* vn_w2     = (const float*)d_in[18];
  const float* vn_b2     = (const float*)d_in[19];
  const float* vn_bn2g   = (const float*)d_in[20];
  const float* vn_bn2b   = (const float*)d_in[21];
  const float* head_w    = (const float*)d_in[22];
  const float* head_b    = (const float*)d_in[23];
  float* out = (float*)d_out;

  // ---- workspace carve (~143 MB) ----
  char* p = (char*)d_ws;
  auto carve = [&](size_t bytes) -> char* {
    char* r = p;
    p += (bytes + 255) & ~(size_t)255;
    return r;
  };
  float* hb       = (float*)carve((size_t)Nn * Dd * 4);   // h + vn[batch]
  float* agg      = (float*)carve((size_t)Nn * Dd * 4);   // also reused as h_next
  bf16*  Abf      = (bf16*)carve((size_t)Nn * K1p * 2);   // GEMM1 activations
  bf16*  A2       = (bf16*)carve((size_t)Nn * N1p * 2);   // GEMM2 activations
  bf16*  W1p      = (bf16*)carve((size_t)Ll * PACKSZ * 2);
  bf16*  W2p      = (bf16*)carve((size_t)Ll * PACKSZ * 2);
  bf16*  V1p      = (bf16*)carve((size_t)(Ll - 1) * PACKSZ * 2);
  bf16*  V2p      = (bf16*)carve((size_t)(Ll - 1) * PACKSZ * 2);
  float* vnA      = (float*)carve((size_t)Gg * Dd * 4);
  float* vnB      = (float*)carve((size_t)Gg * Dd * 4);
  float* pooled   = (float*)carve((size_t)Gg * Dd * 4);
  bf16*  pooledbf = (bf16*)carve((size_t)Gg * K1p * 2);
  bf16*  Tbuf     = (bf16*)carve((size_t)Gg * N1p * 2);
  float* poolH    = (float*)carve((size_t)Gg * Dd * 4);
  float* counts   = (float*)carve((size_t)Gg * 4);
  if ((size_t)(p - (char*)d_ws) > ws_size) return;  // insufficient scratch

  const int* src = eidx;
  const int* dst = eidx + Ee;

  // ---- repack all weights to bf16 WMMA fragment layout ----
  const int packGrid = cdiv(PACKSZ, 256);
  for (int l = 0; l < Ll; ++l) {
    k_pack<<<packGrid, 256, 0, stream>>>(conv_w1 + (size_t)l * K1 * N1, K1, N1, KT1,
                                         PACKSZ, W1p + (size_t)l * PACKSZ);
    k_pack<<<packGrid, 256, 0, stream>>>(conv_w2 + (size_t)l * K2 * N2, K2, N2, KT2,
                                         PACKSZ, W2p + (size_t)l * PACKSZ);
  }
  for (int l = 0; l < Ll - 1; ++l) {
    k_pack<<<packGrid, 256, 0, stream>>>(vn_w1 + (size_t)l * K1 * N1, K1, N1, KT1,
                                         PACKSZ, V1p + (size_t)l * PACKSZ);
    k_pack<<<packGrid, 256, 0, stream>>>(vn_w2 + (size_t)l * K2 * N2, K2, N2, KT2,
                                         PACKSZ, V2p + (size_t)l * PACKSZ);
  }

  k_init_vn<<<cdiv(Gg * Dd, 256), 256, 0, stream>>>(vn_emb, vnA, Gg * Dd);

  const dim3 g1(cdiv(Nn, 128), N1p / 64);   // (235, 10)
  const dim3 g2(cdiv(Nn, 128), N2p / 64);   // (235, 5)
  const dim3 gv1(cdiv(Gg, 128), N1p / 64);  // (8, 10)
  const dim3 gv2(cdiv(Gg, 128), N2p / 64);  // (8, 5)
  const int nodeGrid = cdiv(Nn * 75, 256);
  const int edgeGrid = cdiv(Ee * 75, 256);
  const int makeGrid = cdiv(Nn * 80, 256);

  const float* hcur = x;
  float* vncur = vnA;
  float* vnnext = vnB;

  for (int l = 0; l < Ll; ++l) {
    k_vn_add<<<nodeGrid, 256, 0, stream>>>(hcur, vncur, batch, hb);
    (void)hipMemsetAsync(agg, 0, (size_t)Nn * Dd * 4, stream);
    k_edge<<<edgeGrid, 256, 0, stream>>>(hb, edge_attr, src, dst, agg);
    k_make_A1<<<makeGrid, 256, 0, stream>>>(hb, agg, conv_eps + l, Abf);

    gemm_wmma<1><<<g1, 256, 0, stream>>>(
        Abf, Nn, K1p, W1p + (size_t)l * PACKSZ, N1, N1p,
        conv_b1 + (size_t)l * N1, conv_bng + (size_t)l * N1,
        conv_bnb + (size_t)l * N1, 1, A2, nullptr);
    gemm_wmma<0><<<g2, 256, 0, stream>>>(
        A2, Nn, K2p, W2p + (size_t)l * PACKSZ, N2, N2p,
        conv_b2 + (size_t)l * N2, bn_g + (size_t)l * N2,
        bn_b + (size_t)l * N2, (l < Ll - 1) ? 1 : 0, nullptr, agg /* h_next */);

    if (l < Ll - 1) {
      (void)hipMemsetAsync(pooled, 0, (size_t)Gg * Dd * 4, stream);
      k_segsum<<<nodeGrid, 256, 0, stream>>>(hb, batch, pooled);
      k_make_poolA<<<cdiv(Gg * 80, 256), 256, 0, stream>>>(pooled, vncur, pooledbf);
      gemm_wmma<1><<<gv1, 256, 0, stream>>>(
          pooledbf, Gg, K1p, V1p + (size_t)l * PACKSZ, N1, N1p,
          vn_b1 + (size_t)l * N1, vn_bn1g + (size_t)l * N1,
          vn_bn1b + (size_t)l * N1, 1, Tbuf, nullptr);
      gemm_wmma<0><<<gv2, 256, 0, stream>>>(
          Tbuf, Gg, K2p, V2p + (size_t)l * PACKSZ, N2, N2p,
          vn_b2 + (size_t)l * N2, vn_bn2g + (size_t)l * N2,
          vn_bn2b + (size_t)l * N2, 1, nullptr, vnnext);
      float* t = vncur; vncur = vnnext; vnnext = t;
    }
    hcur = agg;  // h_next lives in the agg buffer; re-read before next memset
  }

  // ---- global mean pool + classification head ----
  (void)hipMemsetAsync(poolH, 0, (size_t)Gg * Dd * 4, stream);
  (void)hipMemsetAsync(counts, 0, (size_t)Gg * 4, stream);
  k_segsum<<<nodeGrid, 256, 0, stream>>>(hcur, batch, poolH);
  k_counts<<<cdiv(Nn, 256), 256, 0, stream>>>(batch, counts);
  k_head<<<cdiv(Gg * Cc, 256), 256, 0, stream>>>(poolH, counts, head_w, head_b, out);
}
